// ImprovedSelectiveSSM_50208167690244
// MI455X (gfx1250) — matrix operand order repaired and checked by reading, not compiled, not run on hardware
//
#include <hip/hip_runtime.h>
#include <hip/hip_bf16.h>
#include <math.h>

// ---------------- CDNA5 WMMA types ----------------
typedef __attribute__((ext_vector_type(16))) _Float16 v16h;
typedef __attribute__((ext_vector_type(8)))  _Float16 half8;
typedef __attribute__((ext_vector_type(8)))  float    v8f;
typedef __attribute__((ext_vector_type(4)))  float    f4;
typedef __attribute__((ext_vector_type(4)))  int      i32x4;

#define BDIM 16
#define LDIM 4096
#define DDIM 512
#define SDIM 64
#define ROWS (BDIM * LDIM)        // 65536
#define NCHUNK 64                 // chunks along L
#define CHUNKT (LDIM / NCHUNK)    // 64

// ------------- workspace layout (bytes) -------------
#define WS_WCAT16   0u                       // 256*512 f16 = 262144
#define WS_WOUT16   262144u                  // 512*64  f16 =  65536
#define WS_DT       327680u                  // ROWS*64 f32 = 16777216 (later: cumprod P)
#define WS_BT       17104896u
#define WS_CT       33882112u
#define WS_XP       50659328u                //                (later: h_local)
#define WS_PC       67436544u                // 16*64*64 f32 = 262144
#define WS_HC       67698688u
#define WS_CARRY    67960832u
// total ~68.2 MB

// ---------------- async global->LDS (gfx1250), guarded ----------------
#if __has_builtin(__builtin_amdgcn_global_load_async_to_lds_b128) && \
    __has_builtin(__builtin_amdgcn_s_wait_asynccnt)
#define USE_ASYNC 1
#else
#define USE_ASYNC 0
#endif

#define AS1 __attribute__((address_space(1)))
#define AS3 __attribute__((address_space(3)))

#if USE_ASYNC
__device__ __forceinline__ AS1 i32x4* to_gbl(const void* p) {
  return (AS1 i32x4*)(uintptr_t)p;               // as(1) value == generic value
}
__device__ __forceinline__ AS3 i32x4* to_lds(void* p) {
  return (AS3 i32x4*)(uint32_t)(uintptr_t)p;     // low 32 bits = LDS offset
}
#endif

// copy 'n128' 16B packets per thread from global src to LDS dst
__device__ __forceinline__ void stage_async(const _Float16* src, _Float16* dst, int n128) {
#if USE_ASYNC
  for (int j = 0; j < n128; ++j)
    __builtin_amdgcn_global_load_async_to_lds_b128(to_gbl(src + j * 8),
                                                   to_lds(dst + j * 8), 0, 0);
#else
  for (int j = 0; j < n128; ++j)
    *(half8*)(dst + j * 8) = *(const half8*)(src + j * 8);
#endif
}

__device__ __forceinline__ v16h make_frag(half8 lo, half8 hi) {
  return __builtin_shufflevector(lo, hi, 0,1,2,3,4,5,6,7,8,9,10,11,12,13,14,15);
}

// ================= k0: convert weights to f16 =================
__global__ void k0_prep(const float* __restrict__ Wdt, const float* __restrict__ WB,
                        const float* __restrict__ WC, const float* __restrict__ Wx,
                        const float* __restrict__ Wout,
                        _Float16* __restrict__ Wcat16, _Float16* __restrict__ Wout16) {
  int i = blockIdx.x * blockDim.x + threadIdx.x;
  int total = 256 * 512 + 512 * 64;
  for (; i < total; i += gridDim.x * blockDim.x) {
    if (i < 256 * 512) {
      int n = i >> 9, k = i & 511;
      int which = n >> 6, s = n & 63;
      const float* src = (which == 0) ? Wdt : (which == 1) ? WB : (which == 2) ? WC : Wx;
      Wcat16[n * 512 + k] = (_Float16)src[s * 512 + k];
    } else {
      int j = i - 256 * 512;
      Wout16[j] = (_Float16)Wout[j];
    }
  }
}

// ================= k1: fused LayerNorm + 4 projections (N=256 GEMM) =================
// block: 256 threads (8 waves), 64 rows per block; K = 512 in chunks of 32.
// B tile is double-buffered and staged with async global->LDS, overlapped with WMMA.
#define AK 520          // 512 + 8 f16 pad
#define BK 40           // 32 + 8 f16 pad
#define BSTRIDE (256 * BK)
__global__ void k1_ln_proj(const float* __restrict__ x,
                           const float* __restrict__ lnw, const float* __restrict__ lnb,
                           const _Float16* __restrict__ Wcat16,
                           const float* __restrict__ bdt, const float* __restrict__ bx,
                           float* __restrict__ dtA, float* __restrict__ BtA,
                           float* __restrict__ CtA, float* __restrict__ xpA) {
  extern __shared__ char smem[];
  _Float16* Atile = (_Float16*)smem;                    // 64 x AK
  _Float16* Btile = (_Float16*)(smem + 64 * AK * 2);    // 2 x 256 x BK (double buffer)

  const int tid  = threadIdx.x;
  const int wave = tid >> 5;
  const int lane = tid & 31;
  const int grp  = lane >> 4;          // 0 or 1 (lane group)
  const int l15  = lane & 15;
  const long rowBase = (long)blockIdx.x * 64;

  // prologue: async-stage B chunk 0 into buffer 0 (overlaps with the LN phase)
  stage_async(Wcat16 + tid * 512, Btile + tid * BK, 4);

  // ---- Phase 1: LayerNorm 8 rows per wave, write f16 into Atile ----
  for (int i = 0; i < 8; ++i) {
    const int row = wave * 8 + i;
    const float* xr = x + (rowBase + row) * DDIM;
    f4 v[4];
    float sum = 0.f, sq = 0.f;
    #pragma unroll
    for (int c = 0; c < 4; ++c) {
      v[c] = ((const f4*)xr)[lane + c * 32];
      #pragma unroll
      for (int j = 0; j < 4; ++j) { sum += v[c][j]; sq += v[c][j] * v[c][j]; }
    }
    #pragma unroll
    for (int off = 16; off > 0; off >>= 1) {
      sum += __shfl_xor(sum, off, 32);
      sq  += __shfl_xor(sq,  off, 32);
    }
    const float mu   = sum * (1.0f / DDIM);
    const float var  = sq * (1.0f / DDIM) - mu * mu;
    const float rstd = rsqrtf(var + 1e-5f);
    #pragma unroll
    for (int c = 0; c < 4; ++c) {
      const int col0 = (lane + c * 32) * 4;
      f4 w = ((const f4*)lnw)[lane + c * 32];
      f4 b = ((const f4*)lnb)[lane + c * 32];
      #pragma unroll
      for (int j = 0; j < 4; ++j)
        Atile[row * AK + col0 + j] = (_Float16)((v[c][j] - mu) * rstd * w[j] + b[j]);
    }
  }

  // ---- Phase 2: WMMA GEMM, M=64 (4 tiles) x N=256 (2 tiles/wave) ----
  v8f acc[4][2];
  const v8f vzero = {0.f,0.f,0.f,0.f,0.f,0.f,0.f,0.f};
  #pragma unroll
  for (int m = 0; m < 4; ++m)
    #pragma unroll
    for (int n = 0; n < 2; ++n) acc[m][n] = vzero;

  for (int kc = 0; kc < 16; ++kc) {
    _Float16* cur = Btile + (kc & 1) * BSTRIDE;
    if (kc < 15) {
      // async-prefetch next K-chunk into the other buffer (its readers finished
      // at the end-of-iteration barrier of kc-1)
      stage_async(Wcat16 + tid * 512 + (kc + 1) * 32,
                  Btile + ((kc + 1) & 1) * BSTRIDE + tid * BK, 4);
#if USE_ASYNC
      __builtin_amdgcn_s_wait_asynccnt(4);  // drain buffer kc (oldest 4, in-order)
#endif
    } else {
#if USE_ASYNC
      __builtin_amdgcn_s_wait_asynccnt(0);
#endif
    }
    __syncthreads();

    // A fragments (16x32 f16 layout: lane group k-split)
    v16h aF[4];
    #pragma unroll
    for (int m = 0; m < 4; ++m) {
      const int row = m * 16 + l15;
      const int kb  = kc * 32 + grp * 8;
      half8 lo = *(const half8*)(Atile + row * AK + kb);
      half8 hi = *(const half8*)(Atile + row * AK + kb + 16);
      aF[m] = make_frag(lo, hi);
    }
    // B fragments (32x16: lane = column, lane group k-split like A)
    v16h bF[2];
    #pragma unroll
    for (int n = 0; n < 2; ++n) {
      const int ncol = (wave * 2 + n) * 16 + l15;
      half8 lo = *(const half8*)(cur + ncol * BK + grp * 8);
      half8 hi = *(const half8*)(cur + ncol * BK + grp * 8 + 16);
      bF[n] = make_frag(lo, hi);
    }
    #pragma unroll
    for (int m = 0; m < 4; ++m)
      #pragma unroll
      for (int n = 0; n < 2; ++n)
        acc[m][n] = __builtin_amdgcn_wmma_f32_16x16x32_f16(
            false, aF[m], false, bF[n], (short)0, acc[m][n], false, false);
    __syncthreads();   // all waves done reading 'cur' before it is restaged
  }

  // ---- Epilogue: scatter the 4 projections (with biases) ----
  #pragma unroll
  for (int m = 0; m < 4; ++m) {
    #pragma unroll
    for (int n = 0; n < 2; ++n) {
      const int col   = (wave * 2 + n) * 16 + l15;
      const int which = col >> 6;
      const int s     = col & 63;
      float* dst = (which == 0) ? dtA : (which == 1) ? BtA : (which == 2) ? CtA : xpA;
      const float bias = (which == 0) ? bdt[s] : (which == 3) ? bx[s] : 0.f;
      #pragma unroll
      for (int v = 0; v < 8; ++v) {
        const long row = rowBase + m * 16 + v + grp * 8;
        dst[row * SDIM + s] = acc[m][n][v] + bias;
      }
    }
  }
}

// ================= k2a: gating + intra-chunk scan =================
// grid (NCHUNK, BDIM), 64 threads (thread = state s). In-place: xpA <- h_local, dtA <- cumprod a.
__global__ void k2a_scan_local(float* __restrict__ dtA, const float* __restrict__ BtA,
                               float* __restrict__ xpA, const float* __restrict__ A_log,
                               float* __restrict__ Pc, float* __restrict__ Hc) {
  const int s = threadIdx.x;
  const int chunk = blockIdx.x, b = blockIdx.y;
  const float A = -__expf(A_log[s]);
  float P = 1.0f, H = 0.0f;
  const long base = ((long)b * LDIM + (long)chunk * CHUNKT);
  for (int t = 0; t < CHUNKT; ++t) {
    const long idx = (base + t) * SDIM + s;
    const float dt = dtA[idx];
    const float Bt = BtA[idx];
    const float xp = xpA[idx];
    const float sp = (dt > 20.f) ? dt : log1pf(__expf(dt));
    const float delta = sp * 0.01f + 1e-4f;
    float da = delta * A;
    da = fminf(fmaxf(da, -10.f), -1e-4f);
    float a = __expf(da);
    a = fminf(fmaxf(a, 0.001f), 0.999f);          // A_safe clip of the reference scan
    const float u = delta * Bt * xp;
    H = a * (H + u);                              // matches log-cumsum algebra
    P *= a;
    xpA[idx] = H;
    dtA[idx] = P;
  }
  Pc[((long)b * NCHUNK + chunk) * SDIM + s] = P;
  Hc[((long)b * NCHUNK + chunk) * SDIM + s] = H;
}

// ================= k2b: cross-chunk scan (tiny) =================
__global__ void k2b_scan_chunks(const float* __restrict__ Pc, const float* __restrict__ Hc,
                                float* __restrict__ carry) {
  const int tid = blockIdx.x * blockDim.x + threadIdx.x;   // 1024 = 16b x 64s
  const int b = tid >> 6, s = tid & 63;
  float c = 0.0f;
  for (int ch = 0; ch < NCHUNK; ++ch) {
    const long i = ((long)b * NCHUNK + ch) * SDIM + s;
    carry[i] = c;
    c = Pc[i] * c + Hc[i];
  }
}

// ================= k3: fix-up + y = (C*h) @ Wout^T + bout + clip(Dp)*x =================
// block: 256 threads (8 waves), 32 rows; K = 64 in 2 chunks of 32; N = 512 (4 tiles/wave).
#define AK3 72
#define WK3 72
__global__ void k3_out_gemm(const float* __restrict__ dtA /*cumprod*/,
                            const float* __restrict__ CtA,
                            const float* __restrict__ xpA /*h_local*/,
                            const float* __restrict__ carry,
                            const _Float16* __restrict__ Wout16,
                            const float* __restrict__ bout, const float* __restrict__ Dp,
                            const float* __restrict__ x, float* __restrict__ out) {
  extern __shared__ char smem[];
  _Float16* Atile = (_Float16*)smem;                    // 32 x AK3
  _Float16* Wlds  = (_Float16*)(smem + 32 * AK3 * 2);   // 512 x WK3

  const int tid  = threadIdx.x;
  const int wave = tid >> 5;
  const int lane = tid & 31;
  const int grp  = lane >> 4;
  const int l15  = lane & 15;
  const long rowBase = (long)blockIdx.x * 32;

  // stage Wout^T columns (LDS row d holds Wout[d][0..63], contiguous in memory)
  #pragma unroll
  for (int r = 0; r < 2; ++r) {
    const int d = tid * 2 + r;
    stage_async(Wout16 + d * 64, Wlds + d * WK3, 8);
  }
#if USE_ASYNC
  __builtin_amdgcn_s_wait_asynccnt(0);
#endif

  // build A tile: y_s = C * (h_local + carry * cumprod), f16
  #pragma unroll
  for (int e = 0; e < 8; ++e) {
    const int idx = tid + e * 256;          // 32*64 elements
    const int s = idx & 63, row = idx >> 6;
    const long grow = rowBase + row;
    const int b = (int)(grow >> 12);
    const int t = (int)(grow & 4095);
    const int ch = t / CHUNKT;
    const float c = carry[((long)b * NCHUNK + ch) * SDIM + s];
    const long gi = grow * SDIM + s;
    const float h = xpA[gi] + c * dtA[gi];
    float y = CtA[gi] * h;
    if (!__builtin_isfinite(y)) y = 0.0f;   // reference's non-finite guard
    Atile[row * AK3 + s] = (_Float16)y;
  }
  __syncthreads();

  v8f acc[2][4];
  const v8f vzero = {0.f,0.f,0.f,0.f,0.f,0.f,0.f,0.f};
  #pragma unroll
  for (int m = 0; m < 2; ++m)
    #pragma unroll
    for (int n = 0; n < 4; ++n) acc[m][n] = vzero;

  #pragma unroll
  for (int kc = 0; kc < 2; ++kc) {
    v16h aF[2];
    #pragma unroll
    for (int m = 0; m < 2; ++m) {
      const int row = m * 16 + l15;
      const int kb  = kc * 32 + grp * 8;
      half8 lo = *(const half8*)(Atile + row * AK3 + kb);
      half8 hi = *(const half8*)(Atile + row * AK3 + kb + 16);
      aF[m] = make_frag(lo, hi);
    }
    v16h bF[4];
    #pragma unroll
    for (int n = 0; n < 4; ++n) {
      const int d = (wave * 4 + n) * 16 + l15;
      const int kb = kc * 32 + grp * 8;
      half8 lo = *(const half8*)(Wlds + d * WK3 + kb);
      half8 hi = *(const half8*)(Wlds + d * WK3 + kb + 16);
      bF[n] = make_frag(lo, hi);
    }
    #pragma unroll
    for (int m = 0; m < 2; ++m)
      #pragma unroll
      for (int n = 0; n < 4; ++n)
        acc[m][n] = __builtin_amdgcn_wmma_f32_16x16x32_f16(
            false, aF[m], false, bF[n], (short)0, acc[m][n], false, false);
  }

  // epilogue: + bout + clip(Dp,-2,2) * x
  #pragma unroll
  for (int n = 0; n < 4; ++n) {
    const int d = (wave * 4 + n) * 16 + l15;
    const float bo = bout[d];
    const float ds = fminf(fmaxf(Dp[d], -2.f), 2.f);
    #pragma unroll
    for (int m = 0; m < 2; ++m) {
      #pragma unroll
      for (int v = 0; v < 8; ++v) {
        const long r = rowBase + m * 16 + v + grp * 8;
        out[r * DDIM + d] = acc[m][n][v] + bo + ds * x[r * DDIM + d];
      }
    }
  }
}

// ================= host launcher =================
extern "C" void kernel_launch(void* const* d_in, const int* in_sizes, int n_in,
                              void* d_out, int out_size, void* d_ws, size_t ws_size,
                              hipStream_t stream) {
  const float* x     = (const float*)d_in[0];
  const float* lnw   = (const float*)d_in[1];
  const float* lnb   = (const float*)d_in[2];
  const float* Wx    = (const float*)d_in[3];
  const float* bx    = (const float*)d_in[4];
  const float* Wdt   = (const float*)d_in[5];
  const float* bdt   = (const float*)d_in[6];
  const float* A_log = (const float*)d_in[7];
  const float* WB    = (const float*)d_in[8];
  const float* WC    = (const float*)d_in[9];
  const float* Dp    = (const float*)d_in[10];
  const float* Wout  = (const float*)d_in[11];
  const float* bout  = (const float*)d_in[12];
  float* out = (float*)d_out;

  char* ws = (char*)d_ws;
  _Float16* Wcat16 = (_Float16*)(ws + WS_WCAT16);
  _Float16* Wout16 = (_Float16*)(ws + WS_WOUT16);
  float* dtA   = (float*)(ws + WS_DT);
  float* BtA   = (float*)(ws + WS_BT);
  float* CtA   = (float*)(ws + WS_CT);
  float* xpA   = (float*)(ws + WS_XP);
  float* Pc    = (float*)(ws + WS_PC);
  float* Hc    = (float*)(ws + WS_HC);
  float* carry = (float*)(ws + WS_CARRY);

  // k0: weight conversion
  k0_prep<<<160, 256, 0, stream>>>(Wdt, WB, WC, Wx, Wout, Wcat16, Wout16);

  // k1: LN + 4 projections, 64 rows/block (double-buffered async B staging)
  const size_t lds1 = (size_t)64 * AK * 2 + (size_t)2 * BSTRIDE * 2;  // ~105 KB (WGP has 320 KB)
  k1_ln_proj<<<ROWS / 64, 256, lds1, stream>>>(x, lnw, lnb, Wcat16, bdt, bx,
                                               dtA, BtA, CtA, xpA);

  // k2a: gating + local chunk scans (1024 blocks)
  k2a_scan_local<<<dim3(NCHUNK, BDIM), SDIM, 0, stream>>>(dtA, BtA, xpA, A_log, Pc, Hc);

  // k2b: cross-chunk scan
  k2b_scan_chunks<<<1, 1024, 0, stream>>>(Pc, Hc, carry);

  // k3: fix-up + output GEMM + residual, 32 rows/block
  const size_t lds3 = (size_t)32 * AK3 * 2 + (size_t)512 * WK3 * 2;   // ~76.5 KB
  k3_out_gemm<<<ROWS / 32, 256, lds3, stream>>>(dtA, CtA, xpA, carry, Wout16,
                                                bout, Dp, x, out);
}
